// MPNN_360777253448
// MI455X (gfx1250) — matrix operand order repaired and checked by reading, not compiled, not run on hardware
//
#include <hip/hip_runtime.h>
#include <hip/hip_bf16.h>
#include <math.h>

// ---------------------------------------------------------------------------
// MPNN forward on MI455X (gfx1250, wave32, WMMA + TDM).
//   N=50000 nodes, E=800000 edges, DIM=128, STEPS=4, POSD=3.
// Message GEMM [E,259]x[259,128] and update GEMM [N,256]x[256,128] run on
// v_wmma_f32_16x16x32_bf16 (A/B bf16, C/D f32). The 3-row edge_attr tail of
// msg_W is applied with f32 VALU FMAs. GELU(exact)+LayerNorm fused in-register,
// scatter-mean via global f32 atomics. Weight tiles staged into LDS with the
// Tensor Data Mover (tensor_load_to_lds + s_wait_tensorcnt), with a
// cooperative-copy fallback if the builtin is unavailable.
// ---------------------------------------------------------------------------

#define NN     50000
#define NE     800000
#define DIM    128
#define NSTEP  4
#define LN_EPS 1e-5f
#define TPW    5          // tiles per wave (amortize 64KB LDS weight stage)

typedef __attribute__((ext_vector_type(16))) __bf16 v16bf;
typedef __attribute__((ext_vector_type(8)))  float  v8f;
typedef __attribute__((ext_vector_type(4)))  unsigned int v4u;
typedef __attribute__((ext_vector_type(8)))  int    v8i_;
typedef __attribute__((ext_vector_type(4)))  int    v4i_;

#if defined(__HIP_DEVICE_COMPILE__) && defined(__gfx1250__) && \
    __has_builtin(__builtin_amdgcn_tensor_load_to_lds) &&      \
    __has_builtin(__builtin_amdgcn_s_wait_tensorcnt)
#define HAVE_TDM 1
#endif

union FragAB {
  v16bf v;
  uint4 q[2];
  unsigned short u[16];
};

__device__ __forceinline__ unsigned short f32_to_bf16_rne(float f) {
  union { float f; unsigned int u; } c; c.f = f;
  unsigned int u = c.u;
  u += 0x7fffu + ((u >> 16) & 1u);
  return (unsigned short)(u >> 16);
}

__device__ __forceinline__ float gelu_exact(float x) {
  return 0.5f * x * (1.0f + erff(x * 0.70710678118654752440f));
}

// Stage a 128x256 bf16 tile (64KB, row-contiguous) from global into LDS.
// TDM path: wave 0 issues one TENSOR_LOAD_TO_LDS (EXEC-independent, wave-level)
// and waits on TENSORcnt; caller's __syncthreads() releases the other waves.
__device__ __forceinline__ void stage_weights(unsigned short* sW,
                                              const unsigned short* gsrc,
                                              int tid, int nthreads) {
#if defined(HAVE_TDM)
  if ((tid >> 5) == 0) {
    const unsigned lds_off = (unsigned)(unsigned long long)sW;  // flat LDS aperture: addr[31:0] = LDS byte offset
    const unsigned long long ga = (unsigned long long)gsrc;
    v4u g0;
    g0[0] = 1u;                                         // count=1, user descriptor
    g0[1] = lds_off;                                    // lds_addr
    g0[2] = (unsigned)(ga & 0xFFFFFFFFu);               // global_addr[31:0]
    g0[3] = (unsigned)((ga >> 32) & 0x01FFFFFFu)        // global_addr[56:32]
            | (2u << 30);                               // type=2 ("image")
    v8i_ g1;
    g1[0] = (int)(1u << 16);      // wg_mask=0, data_size=1 (2 bytes)
    g1[1] = (int)(256u << 16);    // tensor_dim0[15:0]=256 in bits[63:48]
    g1[2] = (int)(128u << 16);    // tensor_dim0 hi=0, tensor_dim1[15:0]=128
    g1[3] = (int)(256u << 16);    // tensor_dim1 hi=0, tile_dim0=256
    g1[4] = 128;                  // tile_dim1=128, tile_dim2=0
    g1[5] = 256;                  // tensor_dim0_stride = 256 elements
    g1[6] = 0;
    g1[7] = 0;
    v4i_ gz; gz[0] = 0; gz[1] = 0; gz[2] = 0; gz[3] = 0;
#if defined(__clang_major__) && (__clang_major__ >= 23)
    v8i_ gz8; gz8[0]=0; gz8[1]=0; gz8[2]=0; gz8[3]=0; gz8[4]=0; gz8[5]=0; gz8[6]=0; gz8[7]=0;
    __builtin_amdgcn_tensor_load_to_lds(g0, g1, gz, gz, gz8, 0);
#else
    __builtin_amdgcn_tensor_load_to_lds(g0, g1, gz, gz, 0);
#endif
    __builtin_amdgcn_s_wait_tensorcnt(0);
  }
#else
  const uint4* src = (const uint4*)gsrc;
  uint4* dst = (uint4*)sW;
  for (int i = tid; i < (DIM * 256) / 8; i += nthreads) dst[i] = src[i];
#endif
}

// --------------------------- small utility kernels -------------------------

__global__ void k_copy_f32(const float* __restrict__ src, float* __restrict__ dst, int n) {
  int i = blockIdx.x * blockDim.x + threadIdx.x;
  if (i < n) dst[i] = src[i];
}

__global__ void k_zero_f32(float* __restrict__ p, int n) {
  int i = blockIdx.x * blockDim.x + threadIdx.x;
  if (i < n) p[i] = 0.0f;
}

__global__ void k_counts(const long long* __restrict__ ei, float* __restrict__ counts) {
  int e = blockIdx.x * blockDim.x + threadIdx.x;
  if (e < NE) atomicAdd(&counts[(int)ei[NE + e]], 1.0f);
}

__global__ void k_f32_to_bf16(const float* __restrict__ src, unsigned short* __restrict__ dst, int n) {
  int i = blockIdx.x * blockDim.x + threadIdx.x;
  if (i < n) dst[i] = f32_to_bf16_rne(src[i]);
}

// Wt[s][n][k] = bf16(W[s][k][n]), k < 256 (kstride = row pitch of W's K dim)
__global__ void k_transposeW(const float* __restrict__ W, unsigned short* __restrict__ Wt,
                             int kstride) {
  int i = blockIdx.x * blockDim.x + threadIdx.x;
  if (i >= NSTEP * 256 * DIM) return;
  int s = i / (256 * DIM);
  int r = i - s * (256 * DIM);
  int k = r / DIM;
  int n = r - k * DIM;
  Wt[(s * DIM + n) * 256 + k] = f32_to_bf16_rne(W[(s * kstride + k) * DIM + n]);
}

// aggbf[i] = bf16(agg[i] / max(counts[node],1))
__global__ void k_agg_scale(const float* __restrict__ agg, const float* __restrict__ counts,
                            unsigned short* __restrict__ aggbf) {
  int i = blockIdx.x * blockDim.x + threadIdx.x;
  if (i >= NN * DIM) return;
  int node = i >> 7;
  float inv = 1.0f / fmaxf(counts[node], 1.0f);
  aggbf[i] = f32_to_bf16_rne(agg[i] * inv);
}

// ------------------------------ message kernel -----------------------------
// grid: E/16/(8*TPW) = 1250 blocks, 256 threads (8 waves), TPW 16-edge tiles
// per wave. LDS: 64KB staged Wt (bf16, [n][k] layout), reused by 640 edges.

__global__ __launch_bounds__(256)
void k_message(const unsigned short* __restrict__ hbf,       // [N][128] bf16
               const unsigned short* __restrict__ WtMsg,     // [128][256] bf16 (transposed)
               const float* __restrict__ msgW_s,             // msg_W[s] base (259x128 f32), tail rows
               const float* __restrict__ bias,               // msg_b[s] [128]
               const float* __restrict__ gamma,              // msg_g[s]
               const float* __restrict__ beta,               // msg_be[s]
               const long long* __restrict__ ei,             // [2][E]
               const float* __restrict__ pos,                // [N][3]
               float* __restrict__ agg)                      // [N][128] f32 accum
{
  extern __shared__ unsigned short sW[];  // 128*256 bf16 = 64KB

  stage_weights(sW, WtMsg, threadIdx.x, blockDim.x);
  __syncthreads();

  const int wave  = threadIdx.x >> 5;
  const int lane  = threadIdx.x & 31;
  const int lm    = lane & 15;
  const int half  = lane >> 4;
  const int rbase = half * 8;
  const int tileBase = (blockIdx.x * (blockDim.x >> 5) + wave) * TPW;

  for (int it = 0; it < TPW; ++it) {
    const int tile = tileBase + it;
    const int e0 = tile * 16;
    if (e0 >= NE) break;

    // Per-lane edge (lanes 16..31 mirror 0..15).
    const int eL = e0 + lm;
    __builtin_prefetch(ei + eL + 256, 0, 0);   // global_prefetch_b8, speculative
    const int r = (int)ei[eL];
    const int c = (int)ei[NE + eL];
    const float ea0 = pos[r * 3 + 0] - pos[c * 3 + 0];
    const float ea1 = pos[r * 3 + 1] - pos[c * 3 + 1];
    const float ea2 = pos[r * 3 + 2] - pos[c * 3 + 2];

    // A fragments straight from global bf16 h: 2x global_load_b128 each.
    // ISA 16-bit A 16x32 layout: lanes 0-15 V0..3=K(kb..kb+7), V4..7=K(kb+16..kb+23).
    FragAB A[8];
#pragma unroll
    for (int k = 0; k < 8; ++k) {
      const unsigned short* rowp = (k < 4) ? (hbf + (size_t)r * DIM) : (hbf + (size_t)c * DIM);
      const int kb = (k & 3) * 32 + half * 8;
      const uint4* p = (const uint4*)(rowp + kb);
      A[k].q[0] = p[0];
      A[k].q[1] = p[2];   // +16 bf16 elements
    }

    // 8 column tiles x 8 k-steps of v_wmma_f32_16x16x32_bf16.
    v8f acc[8];
#pragma unroll
    for (int t = 0; t < 8; ++t) {
      v8f cacc = {};
#pragma unroll
      for (int k = 0; k < 8; ++k) {
        FragAB B;
        // B 16-bit 32x16: lanes 0-15 K=0..7,16..23, lanes 16-31 K=8..15,24..31, col n=lane%16.
        const uint4* p = (const uint4*)(sW + (size_t)(t * 16 + lm) * 256 + k * 32 + half * 8);
        B.q[0] = p[0];
        B.q[1] = p[2];
        cacc = __builtin_amdgcn_wmma_f32_16x16x32_bf16(false, A[k].v, false, B.v,
                                                       (short)0, cacc, false, false);
      }
      acc[t] = cacc;
    }

    // Tail (edge_attr @ Wc) + bias + exact GELU; accumulate LN moments per row.
    float rs1[8] = {0,0,0,0,0,0,0,0};
    float rs2[8] = {0,0,0,0,0,0,0,0};
#pragma unroll
    for (int t = 0; t < 8; ++t) {
      const int n = t * 16 + lm;
      const float wc0 = msgW_s[256 * DIM + n];
      const float wc1 = msgW_s[257 * DIM + n];
      const float wc2 = msgW_s[258 * DIM + n];
      const float b = bias[n];
#pragma unroll
      for (int j = 0; j < 8; ++j) {
        const float a0 = __shfl(ea0, rbase + j, 32);
        const float a1 = __shfl(ea1, rbase + j, 32);
        const float a2 = __shfl(ea2, rbase + j, 32);
        float xv = acc[t][j] + b + a0 * wc0 + a1 * wc1 + a2 * wc2;
        const float g = gelu_exact(xv);
        acc[t][j] = g;
        rs1[j] += g;
        rs2[j] += g * g;
      }
    }
    // Cross-lane butterfly inside the 16-lane half (wave32: masks stay in-half).
#pragma unroll
    for (int m = 1; m < 16; m <<= 1) {
#pragma unroll
      for (int j = 0; j < 8; ++j) {
        rs1[j] += __shfl_xor(rs1[j], m, 32);
        rs2[j] += __shfl_xor(rs2[j], m, 32);
      }
    }
    float mean[8], rstd[8];
#pragma unroll
    for (int j = 0; j < 8; ++j) {
      mean[j] = rs1[j] * (1.0f / DIM);
      const float var = rs2[j] * (1.0f / DIM) - mean[j] * mean[j];
      rstd[j] = rsqrtf(var + LN_EPS);
    }

    // LayerNorm affine + scatter-add into agg (f32 atomics, L2-resident).
#pragma unroll
    for (int t = 0; t < 8; ++t) {
      const int n = t * 16 + lm;
      const float gm = gamma[n];
      const float bt = beta[n];
#pragma unroll
      for (int j = 0; j < 8; ++j) {
        const int cc = __shfl(c, rbase + j, 32);
        const float y = (acc[t][j] - mean[j]) * rstd[j] * gm + bt;
        atomicAdd(&agg[(size_t)cc * DIM + n], y);
      }
    }
  }
}

// ------------------------------- update kernel -----------------------------
// u_in = [h, agg/denom] (both bf16 in ws), GEMM 16x256x128 per node tile,
// bias + LayerNorm + residual into f32 h.

__global__ __launch_bounds__(256)
void k_update(const unsigned short* __restrict__ hbf,     // [N][128] bf16
              const unsigned short* __restrict__ aggbf,   // [N][128] bf16 (scaled)
              const unsigned short* __restrict__ WtUpd,   // [128][256] bf16 transposed
              const float* __restrict__ bias,
              const float* __restrict__ gamma,
              const float* __restrict__ beta,
              float* __restrict__ h)                      // [N][128] f32, updated in place
{
  extern __shared__ unsigned short sW[];  // 64KB

  stage_weights(sW, WtUpd, threadIdx.x, blockDim.x);
  __syncthreads();

  const int wave  = threadIdx.x >> 5;
  const int lane  = threadIdx.x & 31;
  const int lm    = lane & 15;
  const int half  = lane >> 4;
  const int rbase = half * 8;
  const int tileBase = (blockIdx.x * (blockDim.x >> 5) + wave) * TPW;

  for (int it = 0; it < TPW; ++it) {
    const int tile = tileBase + it;
    if (tile * 16 >= NN) break;
    const int node = tile * 16 + lm;   // row this lane supplies for A

    FragAB A[8];
#pragma unroll
    for (int k = 0; k < 8; ++k) {
      const unsigned short* rowp =
          (k < 4) ? (hbf + (size_t)node * DIM) : (aggbf + (size_t)node * DIM);
      const int kb = (k & 3) * 32 + half * 8;
      const uint4* p = (const uint4*)(rowp + kb);
      A[k].q[0] = p[0];
      A[k].q[1] = p[2];
    }

    v8f acc[8];
#pragma unroll
    for (int t = 0; t < 8; ++t) {
      v8f cacc = {};
#pragma unroll
      for (int k = 0; k < 8; ++k) {
        FragAB B;
        const uint4* p = (const uint4*)(sW + (size_t)(t * 16 + lm) * 256 + k * 32 + half * 8);
        B.q[0] = p[0];
        B.q[1] = p[2];
        cacc = __builtin_amdgcn_wmma_f32_16x16x32_bf16(false, A[k].v, false, B.v,
                                                       (short)0, cacc, false, false);
      }
      acc[t] = cacc;
    }

    // bias + LN moments
    float rs1[8] = {0,0,0,0,0,0,0,0};
    float rs2[8] = {0,0,0,0,0,0,0,0};
#pragma unroll
    for (int t = 0; t < 8; ++t) {
      const int n = t * 16 + lm;
      const float b = bias[n];
#pragma unroll
      for (int j = 0; j < 8; ++j) {
        const float xv = acc[t][j] + b;
        acc[t][j] = xv;
        rs1[j] += xv;
        rs2[j] += xv * xv;
      }
    }
#pragma unroll
    for (int m = 1; m < 16; m <<= 1) {
#pragma unroll
      for (int j = 0; j < 8; ++j) {
        rs1[j] += __shfl_xor(rs1[j], m, 32);
        rs2[j] += __shfl_xor(rs2[j], m, 32);
      }
    }
    float mean[8], rstd[8];
#pragma unroll
    for (int j = 0; j < 8; ++j) {
      mean[j] = rs1[j] * (1.0f / DIM);
      const float var = rs2[j] * (1.0f / DIM) - mean[j] * mean[j];
      rstd[j] = rsqrtf(var + LN_EPS);
    }

    // residual: h += LN(...) * g + b  (each element owned by exactly one lane)
#pragma unroll
    for (int t = 0; t < 8; ++t) {
      const int n = t * 16 + lm;
      const float gm = gamma[n];
      const float bt = beta[n];
#pragma unroll
      for (int j = 0; j < 8; ++j) {
        const size_t idx = (size_t)(tile * 16 + rbase + j) * DIM + n;
        const float y = (acc[t][j] - mean[j]) * rstd[j] * gm + bt;
        h[idx] = h[idx] + y;
      }
    }
  }
}

// ------------------------------- launcher ----------------------------------

extern "C" void kernel_launch(void* const* d_in, const int* in_sizes, int n_in,
                              void* d_out, int out_size, void* d_ws, size_t ws_size,
                              hipStream_t stream) {
  const float*      x      = (const float*)d_in[0];
  const float*      pos    = (const float*)d_in[1];
  const long long*  ei     = (const long long*)d_in[2];
  const float*      msg_W  = (const float*)d_in[3];   // [4][259][128]
  const float*      msg_b  = (const float*)d_in[4];   // [4][128]
  const float*      msg_g  = (const float*)d_in[5];
  const float*      msg_be = (const float*)d_in[6];
  const float*      upd_W  = (const float*)d_in[7];   // [4][256][128]
  const float*      upd_b  = (const float*)d_in[8];
  const float*      upd_g  = (const float*)d_in[9];
  const float*      upd_be = (const float*)d_in[10];
  float* h = (float*)d_out;                           // [N][128]

  char* ws = (char*)d_ws;
  size_t off = 0;
  unsigned short* hbf   = (unsigned short*)(ws + off); off += (size_t)NN * DIM * 2;   // 12.8 MB
  unsigned short* aggbf = (unsigned short*)(ws + off); off += (size_t)NN * DIM * 2;   // 12.8 MB
  float*          agg   = (float*)(ws + off);          off += (size_t)NN * DIM * 4;   // 25.6 MB
  float*          counts= (float*)(ws + off);          off += (size_t)NN * 4;         // 0.2 MB
  unsigned short* WtMsg = (unsigned short*)(ws + off); off += (size_t)NSTEP * DIM * 256 * 2;
  unsigned short* WtUpd = (unsigned short*)(ws + off); off += (size_t)NSTEP * DIM * 256 * 2;

  const int nhd = NN * DIM;

  // One-time prep (re-done each call for determinism).
  k_copy_f32<<<(nhd + 255) / 256, 256, 0, stream>>>(x, h, nhd);
  k_zero_f32<<<(NN + 255) / 256, 256, 0, stream>>>(counts, NN);
  k_counts<<<(NE + 255) / 256, 256, 0, stream>>>(ei, counts);
  k_transposeW<<<(NSTEP * 256 * DIM + 255) / 256, 256, 0, stream>>>(msg_W, WtMsg, 259);
  k_transposeW<<<(NSTEP * 256 * DIM + 255) / 256, 256, 0, stream>>>(upd_W, WtUpd, 256);

  const size_t shbytes = (size_t)DIM * 256 * 2;     // 64 KB of LDS (WGP has 320 KB)
  const int msg_blocks = NE / 16 / (8 * TPW);       // 1250
  const int upd_blocks = (NN / 16 + 8 * TPW - 1) / (8 * TPW);  // 79

  for (int s = 0; s < NSTEP; ++s) {
    k_f32_to_bf16<<<(nhd + 255) / 256, 256, 0, stream>>>(h, hbf, nhd);
    k_zero_f32<<<(nhd + 255) / 256, 256, 0, stream>>>(agg, nhd);
    k_message<<<msg_blocks, 256, shbytes, stream>>>(
        hbf, WtMsg + (size_t)s * DIM * 256, msg_W + (size_t)s * 259 * DIM,
        msg_b + s * DIM, msg_g + s * DIM, msg_be + s * DIM, ei, pos, agg);
    k_agg_scale<<<(nhd + 255) / 256, 256, 0, stream>>>(agg, counts, aggbf);
    k_update<<<upd_blocks, 256, shbytes, stream>>>(
        hbf, aggbf, WtUpd + (size_t)s * DIM * 256,
        upd_b + s * DIM, upd_g + s * DIM, upd_be + s * DIM, h);
  }
}
